// MultiHeadAttention_81801947120247
// MI455X (gfx1250) — compile-verified, matrix operand order checked by reading, not run on hardware
//
#include <hip/hip_runtime.h>

// ---------------------------------------------------------------------------
// MI455X (gfx1250) fused multi-head attention.
// bf16 WMMA + fp32 accumulate; async global->LDS (ASYNCcnt) double buffering.
// ---------------------------------------------------------------------------

typedef __bf16 bf16;
typedef __attribute__((ext_vector_type(16))) __bf16 v16bf;
typedef __attribute__((ext_vector_type(8)))  float  v8f;
typedef int v4i __attribute__((vector_size(16)));

#define B_  2
#define S_  2048
#define D_  1024
#define H_  16
#define HD_ 64

#define AS_GLOBAL __attribute__((address_space(1)))
#define AS_LDS    __attribute__((address_space(3)))

union FragBF { uint4 u[2]; v16bf v; };

// ---- gfx1250 async global->LDS path (guarded; sync fallback) --------------
#if __has_builtin(__builtin_amdgcn_global_load_async_to_lds_b128)
  #define HAS_ASYNC 1
#else
  #define HAS_ASYNC 0
#endif

__device__ __forceinline__ void async_copy_b128(const bf16* gsrc, bf16* ldst) {
#if HAS_ASYNC
    __builtin_amdgcn_global_load_async_to_lds_b128(
        (AS_GLOBAL v4i*)gsrc,
        (AS_LDS v4i*)ldst,
        0, 0);
#else
    *(uint4*)ldst = *(const uint4*)gsrc;
#endif
}

#if HAS_ASYNC
  #if __has_builtin(__builtin_amdgcn_s_wait_asynccnt)
    #define WAIT_ASYNC(n) __builtin_amdgcn_s_wait_asynccnt(n)
  #else
    #define WAIT_ASYNC(n) asm volatile("s_wait_asynccnt " #n ::: "memory")
  #endif
#else
  #define WAIT_ASYNC(n) do { } while (0)
#endif

__device__ __forceinline__ v8f wmma_bf16(v16bf a, v16bf b, v8f c) {
    return __builtin_amdgcn_wmma_f32_16x16x32_bf16(false, a, false, b, (short)0, c, false, false);
}

// 16x32 bf16 fragment from a K-contiguous LDS tile (A-layout; NT B-frags load
// identically from the other operand's rows).
__device__ __forceinline__ v16bf load_frag(const bf16* tile, int row, int ld, int kbase, int lane) {
    const int half = (lane >> 4) & 1;
    const bf16* p = tile + row * ld + kbase + (half ? 8 : 0);
    FragBF f;
    f.u[0] = *(const uint4*)p;
    f.u[1] = *(const uint4*)(p + 16);
    return f.v;
}

// ---------------------------------------------------------------------------
// fp32 -> bf16 bulk convert (one-time pass; keeps GEMM hot loops VALU-free)
// ---------------------------------------------------------------------------
__global__ __launch_bounds__(256)
void cvt_f32_bf16(const float* __restrict__ in, bf16* __restrict__ out, int n4) {
    const int i = blockIdx.x * 256 + threadIdx.x;
    if (i < n4) {
        const float4 f = ((const float4*)in)[i];
        union { bf16 b[4]; uint2 u; } o;
        o.b[0] = (bf16)f.x; o.b[1] = (bf16)f.y;
        o.b[2] = (bf16)f.z; o.b[3] = (bf16)f.w;
        ((uint2*)out)[i] = o.u;
    }
}

// ---------------------------------------------------------------------------
// NT GEMM (bf16 x bf16, fp32 acc): Out[m,n] = (sum_k A[m,k]*W[n,k] + bias[n])*scale
// Block 256 thr = 8 waves; tile 128x128; K-slab 64, double-buffered async LDS.
// MODE 0: bf16 out in [B,H,S,HD] layout.  MODE 1: fp32 out row-major [M,D].
// ---------------------------------------------------------------------------
template <int MODE>
__global__ __launch_bounds__(256)
void gemm_nt(const bf16* __restrict__ A, const bf16* __restrict__ W,
             const float* __restrict__ bias, void* __restrict__ Out,
             int M, int N, int K, float out_scale)
{
    __shared__ alignas(16) bf16 As[2][128 * 64];
    __shared__ alignas(16) bf16 Ws[2][128 * 64];

    const int t    = threadIdx.x;
    const int lane = t & 31;
    const int wid  = t >> 5;
    const int bm   = blockIdx.x * 128;
    const int bn   = blockIdx.y * 128;
    const int waveM = (wid & 1) * 64;
    const int waveN = (wid >> 1) * 32;
    const int r    = lane & 15;
    const int half = (lane >> 4) & 1;

    v8f acc[4][2];
    #pragma unroll
    for (int i = 0; i < 4; i++)
        #pragma unroll
        for (int j = 0; j < 2; j++) acc[i][j] = {};

    const bf16* gA = A + (size_t)bm * K;
    const bf16* gW = W + (size_t)bn * K;

    // issue one 128x64 bf16 tile (16KB): 1024 b128 chunks, 4 per thread
    auto issue_tile = [&](const bf16* gbase, bf16* lbase, int k0) {
        #pragma unroll
        for (int i = 0; i < 4; i++) {
            const int chunk = i * 256 + t;
            const int row  = chunk >> 3;
            const int koff = (chunk & 7) * 8;
            async_copy_b128(gbase + (size_t)row * K + k0 + koff, lbase + row * 64 + koff);
        }
    };

    const int KT = K >> 6;                       // 64-wide K slabs
    issue_tile(gA, As[0], 0);
    issue_tile(gW, Ws[0], 0);

    for (int kt = 0; kt < KT; ++kt) {
        const int cur = kt & 1;
        __syncthreads();                          // prior slab reads complete
        if (kt + 1 < KT) {
            issue_tile(gA, As[cur ^ 1], (kt + 1) << 6);
            issue_tile(gW, Ws[cur ^ 1], (kt + 1) << 6);
            WAIT_ASYNC(8);                        // this wave's slab-kt copies done
        } else {
            WAIT_ASYNC(0);
        }
        __syncthreads();                          // all waves' copies landed

        #pragma unroll
        for (int ks = 0; ks < 2; ++ks) {
            const int kb = ks * 32;
            v16bf bfrag[2];
            #pragma unroll
            for (int nj = 0; nj < 2; nj++)
                bfrag[nj] = load_frag(Ws[cur], waveN + nj * 16 + r, 64, kb, lane);
            #pragma unroll
            for (int mi = 0; mi < 4; mi++) {
                v16bf afrag = load_frag(As[cur], waveM + mi * 16 + r, 64, kb, lane);
                #pragma unroll
                for (int nj = 0; nj < 2; nj++)
                    acc[mi][nj] = wmma_bf16(afrag, bfrag[nj], acc[mi][nj]);
            }
        }
    }

    // epilogue: C-layout (m,n): lane = (m%16>=8?16:0)+n%16, vgpr = m%8
    #pragma unroll
    for (int mi = 0; mi < 4; mi++) {
        #pragma unroll
        for (int nj = 0; nj < 2; nj++) {
            const int n = bn + waveN + nj * 16 + r;
            const float bv = bias[n];
            #pragma unroll
            for (int v = 0; v < 8; v++) {
                const int m = bm + waveM + mi * 16 + half * 8 + v;
                const float val = (acc[mi][nj][v] + bv) * out_scale;
                if (MODE == 0) {
                    const int b = m >> 11, s = m & (S_ - 1);
                    const int h = n >> 6,  hd = n & 63;
                    ((bf16*)Out)[((size_t)(b * H_ + h) * S_ + s) * HD_ + hd] = (bf16)val;
                } else {
                    ((float*)Out)[(size_t)m * D_ + n] = val;
                }
            }
        }
    }
}

// ---------------------------------------------------------------------------
// Fused attention, flash-style online softmax (== reference max-subtract
// softmax + renormalize).  q pre-scaled by 1/sqrt(HD).
// Grid: (S/128, B*H); 8 waves; wave owns 16 query rows.
// K tiles: async double-buffered.  V tiles: sync transpose-at-fill.
// ---------------------------------------------------------------------------
__global__ __launch_bounds__(256)
void attn_kernel(const bf16* __restrict__ qh, const bf16* __restrict__ kh,
                 const bf16* __restrict__ vh, bf16* __restrict__ ctx)
{
    __shared__ alignas(16) bf16 Ks[2][64 * 64];   // [key][hd], double buffered
    __shared__ alignas(16) bf16 Vs[64 * 64];      // transposed: [hd][key]
    __shared__ alignas(16) bf16 Ps[8][16 * 64];   // per-wave P tile

    const int t    = threadIdx.x;
    const int lane = t & 31;
    const int wid  = t >> 5;
    const int r    = lane & 15;
    const int half = (lane >> 4) & 1;

    const int bh = blockIdx.y;
    const int b  = bh / H_, h = bh % H_;
    const size_t headBase = (size_t)bh * S_ * HD_;
    const int qrow0 = blockIdx.x * 128 + wid * 16;

    // persistent Q fragments (rows qrow0..+15, hd 0..63)
    v16bf qf[2];
    {
        const bf16* qp = qh + headBase + (size_t)(qrow0 + r) * HD_;
        FragBF f0, f1;
        f0.u[0] = *(const uint4*)(qp +      (half ? 8 : 0));
        f0.u[1] = *(const uint4*)(qp + 16 + (half ? 8 : 0));
        f1.u[0] = *(const uint4*)(qp + 32 + (half ? 8 : 0));
        f1.u[1] = *(const uint4*)(qp + 48 + (half ? 8 : 0));
        qf[0] = f0.v; qf[1] = f1.v;
    }

    // 64x64 bf16 K tile (8KB): 512 b128 chunks, 2 per thread
    auto issue_ktile = [&](int j, bf16* lbase) {
        #pragma unroll
        for (int i = 0; i < 2; i++) {
            const int chunk = i * 256 + t;
            const int row  = chunk >> 3;
            const int koff = (chunk & 7) * 8;
            async_copy_b128(kh + headBase + (size_t)(j + row) * HD_ + koff,
                            lbase + row * 64 + koff);
        }
    };

    float m_run[8], l_run[8];
    #pragma unroll
    for (int v = 0; v < 8; v++) { m_run[v] = -3.0e38f; l_run[v] = 0.0f; }
    v8f octx[4];
    #pragma unroll
    for (int i = 0; i < 4; i++) octx[i] = {};

    const int NT = S_ / 64;
    issue_ktile(0, Ks[0]);

    for (int jt = 0; jt < NT; ++jt) {
        const int j   = jt * 64;
        const int cur = jt & 1;
        __syncthreads();                          // prior tile reads complete
        if (jt + 1 < NT) issue_ktile(j + 64, Ks[cur ^ 1]);

        {   // V tile (sync): transpose into Vs while K DMA is in flight
            const int rowK = t >> 2;
            const int c    = (t & 3) * 16;
            const bf16* gv = vh + headBase + (size_t)(j + rowK) * HD_ + c;
            #pragma unroll
            for (int i = 0; i < 16; i++) Vs[(c + i) * 64 + rowK] = gv[i];
            if (jt + 1 < NT)
                __builtin_prefetch(gv + 64 * HD_, 0, 3);   // global_prefetch next V tile
        }

        if (jt + 1 < NT) { WAIT_ASYNC(2); } else { WAIT_ASYNC(0); }
        __syncthreads();                          // K tile landed + Vs filled

        // scores S[16x64] = Q x K^T  (8 WMMAs)
        v8f sacc[4];
        #pragma unroll
        for (int ns = 0; ns < 4; ns++) {
            v8f a = {};
            a = wmma_bf16(qf[0], load_frag(Ks[cur], ns * 16 + r, 64, 0,  lane), a);
            a = wmma_bf16(qf[1], load_frag(Ks[cur], ns * 16 + r, 64, 32, lane), a);
            sacc[ns] = a;
        }

        // online softmax; rows live across the 16 lanes of each half-wave
        #pragma unroll
        for (int v = 0; v < 8; v++) {
            float mx = fmaxf(fmaxf(sacc[0][v], sacc[1][v]), fmaxf(sacc[2][v], sacc[3][v]));
            mx = fmaxf(mx, __shfl_xor(mx, 8, 32));
            mx = fmaxf(mx, __shfl_xor(mx, 4, 32));
            mx = fmaxf(mx, __shfl_xor(mx, 2, 32));
            mx = fmaxf(mx, __shfl_xor(mx, 1, 32));
            const float mnew = fmaxf(m_run[v], mx);
            const float corr = __expf(m_run[v] - mnew);
            float psum = 0.0f;
            #pragma unroll
            for (int ns = 0; ns < 4; ns++) {
                const float p = __expf(sacc[ns][v] - mnew);
                sacc[ns][v] = p;
                psum += p;
            }
            psum += __shfl_xor(psum, 8, 32);
            psum += __shfl_xor(psum, 4, 32);
            psum += __shfl_xor(psum, 2, 32);
            psum += __shfl_xor(psum, 1, 32);
            l_run[v] = l_run[v] * corr + psum;
            m_run[v] = mnew;
            #pragma unroll
            for (int hs = 0; hs < 4; hs++) octx[hs][v] *= corr;
        }

        // C-layout -> A-fragment layout via per-wave LDS bounce
        bf16* pw = &Ps[wid][0];
        #pragma unroll
        for (int ns = 0; ns < 4; ns++)
            #pragma unroll
            for (int v = 0; v < 8; v++)
                pw[(half * 8 + v) * 64 + ns * 16 + r] = (bf16)sacc[ns][v];

        v16bf pf0 = load_frag(pw, r, 64, 0,  lane);
        v16bf pf1 = load_frag(pw, r, 64, 32, lane);

        // ctx += P x V   (8 WMMAs; V^T tile is K-contiguous)
        #pragma unroll
        for (int hs = 0; hs < 4; hs++) {
            octx[hs] = wmma_bf16(pf0, load_frag(Vs, hs * 16 + r, 64, 0,  lane), octx[hs]);
            octx[hs] = wmma_bf16(pf1, load_frag(Vs, hs * 16 + r, 64, 32, lane), octx[hs]);
        }
    }

    // finalize: divide by running sum; ctx in [B,S,D] bf16 for the out-proj
    #pragma unroll
    for (int hs = 0; hs < 4; hs++)
        #pragma unroll
        for (int v = 0; v < 8; v++) {
            const int srow = qrow0 + half * 8 + v;
            const float val = octx[hs][v] / l_run[v];
            ctx[(size_t)(b * S_ + srow) * D_ + h * HD_ + hs * 16 + r] = (bf16)val;
        }
}

// ---------------------------------------------------------------------------
extern "C" void kernel_launch(void* const* d_in, const int* in_sizes, int n_in,
                              void* d_out, int out_size, void* d_ws, size_t ws_size,
                              hipStream_t stream) {
    const float* Q  = (const float*)d_in[0];
    const float* K  = (const float*)d_in[1];
    const float* V  = (const float*)d_in[2];
    const float* Wq = (const float*)d_in[3];
    const float* bq = (const float*)d_in[4];
    const float* Wk = (const float*)d_in[5];
    const float* bk = (const float*)d_in[6];
    const float* Wv = (const float*)d_in[7];
    const float* bv = (const float*)d_in[8];
    const float* Wo = (const float*)d_in[9];
    const float* bo = (const float*)d_in[10];

    const size_t actElems = (size_t)B_ * S_ * D_;   // 4.19M
    const size_t wElems   = (size_t)D_ * D_;        // 1.05M

    bf16* p = (bf16*)d_ws;
    bf16* Qb  = p; p += actElems;
    bf16* Kb  = p; p += actElems;
    bf16* Vb  = p; p += actElems;
    bf16* Wqb = p; p += wElems;
    bf16* Wkb = p; p += wElems;
    bf16* Wvb = p; p += wElems;
    bf16* Wob = p; p += wElems;
    bf16* qhd = p; p += actElems;
    bf16* khd = p; p += actElems;
    bf16* vhd = p; p += actElems;
    bf16* ctx = p; p += actElems;

    dim3 blk(256);
    const int act4 = (int)(actElems / 4), w4 = (int)(wElems / 4);
    cvt_f32_bf16<<<dim3(act4 / 256), blk, 0, stream>>>(Q, Qb, act4);
    cvt_f32_bf16<<<dim3(act4 / 256), blk, 0, stream>>>(K, Kb, act4);
    cvt_f32_bf16<<<dim3(act4 / 256), blk, 0, stream>>>(V, Vb, act4);
    cvt_f32_bf16<<<dim3(w4 / 256),   blk, 0, stream>>>(Wq, Wqb, w4);
    cvt_f32_bf16<<<dim3(w4 / 256),   blk, 0, stream>>>(Wk, Wkb, w4);
    cvt_f32_bf16<<<dim3(w4 / 256),   blk, 0, stream>>>(Wv, Wvb, w4);
    cvt_f32_bf16<<<dim3(w4 / 256),   blk, 0, stream>>>(Wo, Wob, w4);

    const int M = B_ * S_;
    dim3 gGemm(M / 128, D_ / 128);
    const float qscale = 0.125f;                    // 1/sqrt(HD) folded into q
    gemm_nt<0><<<gGemm, blk, 0, stream>>>(Qb, Wqb, bq, qhd, M, D_, D_, qscale);
    gemm_nt<0><<<gGemm, blk, 0, stream>>>(Kb, Wkb, bk, khd, M, D_, D_, 1.0f);
    gemm_nt<0><<<gGemm, blk, 0, stream>>>(Vb, Wvb, bv, vhd, M, D_, D_, 1.0f);

    dim3 gAttn(S_ / 128, B_ * H_);
    attn_kernel<<<gAttn, blk, 0, stream>>>(qhd, khd, vhd, ctx);

    gemm_nt<1><<<gGemm, blk, 0, stream>>>(ctx, Wob, bo, (float*)d_out, M, D_, D_, 1.0f);
}
